// MMFP4FusedMoE_27333171871990
// MI455X (gfx1250) — matrix operand order repaired and checked by reading, not compiled, not run on hardware
//
#include <hip/hip_runtime.h>

typedef unsigned short u16;
typedef unsigned int   u32;
typedef __attribute__((ext_vector_type(16))) __bf16 v16bf;
typedef __attribute__((ext_vector_type(8)))  float  v8f;

#define TOK  2048
#define NEXP 64
#define TOPK 4
#define DDIM 1024
#define IDIM 768
#define GRP  128
#define CAPE 256

// ---------- helpers ----------
__device__ __forceinline__ u16 f2bf(float f) {
  u32 u = __float_as_uint(f);
  u32 r = u + 0x7FFFu + ((u >> 16) & 1u);   // round-to-nearest-even
  return (u16)(r >> 16);
}

// Branchless fp4 E2M1 dequant: place exp/mant bits at the f32 field positions.
// bits = (c&7)<<22 gives exactly fp4_value * 2^-126 (e==0 lands on f32 denormals,
// which stay enabled: .amdhsa_float_denorm_mode_32 3). One fmul by s*2^126 finishes.
__device__ __forceinline__ void dec2(int pk, float sbig, __bf16& o0, __bf16& o1) {
  u32 c0 = (u32)pk & 0xFu;
  u32 c1 = ((u32)pk >> 4) & 0xFu;
  u32 b0 = ((c0 & 7u) << 22) | ((c0 & 8u) << 28);
  u32 b1 = ((c1 & 7u) << 22) | ((c1 & 8u) << 28);
  o0 = (__bf16)(__uint_as_float(b0) * sbig);
  o1 = (__bf16)(__uint_as_float(b1) * sbig);
}

__device__ __forceinline__ v8f wmma_bf16(v16bf a, v16bf b, v8f c) {
  return __builtin_amdgcn_wmma_f32_16x16x32_bf16(false, a, false, b,
                                                 (short)0, c, false, false);
}

// ---------- fp32 -> bf16 elementwise ----------
__global__ void k_to_bf16(const float* __restrict__ x, u16* __restrict__ xb) {
  int i = (blockIdx.x * 256 + threadIdx.x) * 4;
  float4 v = *(const float4*)(x + i);
  u32 lo = (u32)f2bf(v.x) | ((u32)f2bf(v.y) << 16);
  u32 hi = (u32)f2bf(v.z) | ((u32)f2bf(v.w) << 16);
  *(uint2*)(xb + i) = make_uint2(lo, hi);
}

// ---------- router: logits + top-4 + softmax ----------
__global__ void k_router(const float* __restrict__ x, const float* __restrict__ gw,
                         int* __restrict__ tidx, float* __restrict__ twt) {
  int t = blockIdx.x, e = threadIdx.x;       // blockDim = 64
  const float4* xv = (const float4*)(x + (size_t)t * DDIM);
  const float4* wv = (const float4*)(gw + (size_t)e * DDIM);
  float acc = 0.f;
#pragma unroll 4
  for (int i = 0; i < DDIM / 4; ++i) {
    float4 a = xv[i], b = wv[i];
    acc += a.x * b.x + a.y * b.y + a.z * b.z + a.w * b.w;
  }
  __shared__ float sl[NEXP];
  sl[e] = acc;
  __syncthreads();
  if (e == 0) {
    float loc[NEXP];
    for (int j = 0; j < NEXP; ++j) loc[j] = sl[j];
    int sel[TOPK]; float lv[TOPK];
    for (int k = 0; k < TOPK; ++k) {
      int bi = 0; float bv = loc[0];
      for (int j = 1; j < NEXP; ++j)
        if (loc[j] > bv) { bv = loc[j]; bi = j; }
      sel[k] = bi; lv[k] = bv; loc[bi] = -3.4e38f;
    }
    float m = lv[0], sum = 0.f, w[TOPK];
    for (int k = 0; k < TOPK; ++k) { w[k] = __expf(lv[k] - m); sum += w[k]; }
    float inv = 1.f / sum;
    for (int k = 0; k < TOPK; ++k) {
      tidx[t * TOPK + k] = sel[k];
      twt [t * TOPK + k] = w[k] * inv;
    }
  }
}

// ---------- deterministic position assignment (stable token order) ----------
__global__ void k_assign_pos(const int* __restrict__ tidx, int* __restrict__ pos) {
  int e = blockIdx.x, lane = threadIdx.x;    // 64 blocks x 32 lanes
  int cnt = 0;
  for (int base = 0; base < TOK * TOPK; base += 32) {
    int a = base + lane;
    int me = tidx[a];
    u32 m = (u32)__ballot(me == e);
    if (me == e) {
      int p = cnt + __popc(m & ((1u << lane) - 1u));
      pos[a] = (p < CAPE) ? p : -1;
    }
    cnt += __popc(m);
  }
}

// ---------- dispatch: gather token rows into expert buffers (bf16) ----------
__global__ void k_dispatch(const float* __restrict__ x, const int* __restrict__ tidx,
                           const int* __restrict__ pos, u16* __restrict__ buf) {
  int a = blockIdx.x;
  int p = pos[a];
  if (p < 0) return;
  int e = tidx[a];
  int t = a / TOPK;
  int c = threadIdx.x * 4;
  float4 v = *(const float4*)(x + (size_t)t * DDIM + c);
  u32 lo = (u32)f2bf(v.x) | ((u32)f2bf(v.y) << 16);
  u32 hi = (u32)f2bf(v.z) | ((u32)f2bf(v.w) << 16);
  *(uint2*)(buf + ((size_t)e * CAPE + p) * DDIM + c) = make_uint2(lo, hi);
}

// ---------- tiled WMMA GEMM, fp4 weights dequantized inline ----------
// block = 256 thr = 8 waves as 2(Mgrp) x 4(N); block tile 64(M) x 64(N);
// K staged through LDS in chunks of 128 (== one fp4 scale group).
// SWIGLU=true : D = [silu(A@W0^T) * (A@W1^T)] as bf16 -> outH
// SWIGLU=false: D = A@W0^T as fp32               -> outF
template<int KDIM, bool SWIGLU>
__global__ __launch_bounds__(256) void k_gemm(
    const u16* __restrict__ A,
    const int* __restrict__ p0, const float* __restrict__ s0,
    const int* __restrict__ p1, const float* __restrict__ s1,
    u16* __restrict__ outH, float* __restrict__ outF, int ldo,
    long sA_, long sP, long sS, long sO) {
  constexpr int KG = KDIM / GRP;
  const int tid = threadIdx.x, lane = tid & 31;
  const int wave = tid >> 5, wn = wave & 3, wg = wave >> 2;
  const int m0 = blockIdx.x * 64, n0 = blockIdx.y * 64;
  const long z = blockIdx.z;
  const u16*   Ae  = A + z * sA_;
  const int*   p0e = p0 + z * sP;
  const float* s0e = s0 + z * sS;
  const int*   p1e = SWIGLU ? (p1 + z * sP) : nullptr;
  const float* s1e = SWIGLU ? (s1 + z * sS) : nullptr;

  __shared__ __align__(16) u16 sAt[64 * 128];   // 16 KB

  v8f acc0[2], acc1[2];
#pragma unroll
  for (int fr = 0; fr < 2; ++fr)
#pragma unroll
    for (int i = 0; i < 8; ++i) { acc0[fr][i] = 0.f; acc1[fr][i] = 0.f; }

  const int khalf = lane >> 4;
  const int col   = n0 + wn * 16 + (lane & 15);     // weight row / out column
  const long colP = (long)col * (KDIM / 2);
  const long colS = (long)col * KG;

  const int srow = tid >> 2;                        // staging: row 0..63
  const int scol = (tid & 3) * 32;                  // staging: 32 u16 = 64 B

  for (int kc = 0; kc < KDIM; kc += 128) {
    __syncthreads();
    {
      const uint4* s = (const uint4*)(Ae + (size_t)(m0 + srow) * KDIM + kc + scol);
      uint4* d = (uint4*)&sAt[srow * 128 + scol];
#pragma unroll
      for (int q = 0; q < 4; ++q) d[q] = s[q];
    }
    __syncthreads();

    const float sb0 = s0e[colS + (kc >> 7)] * 0x1p126f;
    const float sb1 = SWIGLU ? s1e[colS + (kc >> 7)] * 0x1p126f : 0.f;

#pragma unroll
    for (int kk = 0; kk < 4; ++kk) {
      const int koff = kk * 32;
      union { v16bf v; uint4 q[2]; } a0, a1;
      {
        const u16* base0 = &sAt[(wg * 32 + (lane & 15)) * 128 + koff + khalf * 16];
        a0.q[0] = *(const uint4*)base0;
        a0.q[1] = *(const uint4*)(base0 + 8);
        const u16* base1 = base0 + 16 * 128;
        a1.q[0] = *(const uint4*)base1;
        a1.q[1] = *(const uint4*)(base1 + 8);
      }
      const long dhalf = colP + ((kc + koff + khalf * 16) >> 1);

      union { v16bf v; __bf16 b[16]; } b0;
      {
        const int* pp = p0e + dhalf;
        __builtin_prefetch(pp + 64, 0, 1);
#pragma unroll
        for (int j = 0; j < 8; ++j) dec2(pp[j], sb0, b0.b[2 * j], b0.b[2 * j + 1]);
      }
      acc0[0] = wmma_bf16(a0.v, b0.v, acc0[0]);
      acc0[1] = wmma_bf16(a1.v, b0.v, acc0[1]);

      if constexpr (SWIGLU) {
        union { v16bf v; __bf16 b[16]; } b1;
        const int* pp = p1e + dhalf;
        __builtin_prefetch(pp + 64, 0, 1);
#pragma unroll
        for (int j = 0; j < 8; ++j) dec2(pp[j], sb1, b1.b[2 * j], b1.b[2 * j + 1]);
        acc1[0] = wmma_bf16(a0.v, b1.v, acc1[0]);
        acc1[1] = wmma_bf16(a1.v, b1.v, acc1[1]);
      }
    }
  }

#pragma unroll
  for (int fr = 0; fr < 2; ++fr) {
    const int orow0 = m0 + wg * 32 + fr * 16 + khalf * 8;
#pragma unroll
    for (int v = 0; v < 8; ++v) {
      if constexpr (SWIGLU) {
        float g = acc0[fr][v], u = acc1[fr][v];
        float h = g * (1.f / (1.f + __expf(-g))) * u;   // silu(g) * u
        (outH + z * sO)[(size_t)(orow0 + v) * ldo + col] = f2bf(h);
      } else {
        (outF + z * sO)[(size_t)(orow0 + v) * ldo + col] = acc0[fr][v];
      }
    }
  }
}

// ---------- weighted combine: out += sum_k w_k * eo[e_k, pos_k, :] ----------
__global__ void k_combine(const int* __restrict__ tidx, const int* __restrict__ pos,
                          const float* __restrict__ twt, const float* __restrict__ eo,
                          float* __restrict__ out) {
  int t = blockIdx.x;
#pragma unroll
  for (int dd = 0; dd < 4; ++dd) {
    int d = threadIdx.x + dd * 256;
    float acc = out[(size_t)t * DDIM + d];   // shared-expert result already here
#pragma unroll
    for (int k = 0; k < TOPK; ++k) {
      int a = t * TOPK + k;
      int p = pos[a];
      if (p >= 0) {
        int e = tidx[a];
        acc += twt[a] * eo[((size_t)e * CAPE + p) * DDIM + d];
      }
    }
    out[(size_t)t * DDIM + d] = acc;
  }
}

// ---------- host launch ----------
extern "C" void kernel_launch(void* const* d_in, const int* in_sizes, int n_in,
                              void* d_out, int out_size, void* d_ws, size_t ws_size,
                              hipStream_t stream) {
  const float* x   = (const float*)d_in[0];
  const float* gw  = (const float*)d_in[1];
  const int*   gp  = (const int*)  d_in[2];
  const float* gsc = (const float*)d_in[3];
  const int*   up  = (const int*)  d_in[4];
  const float* usc = (const float*)d_in[5];
  const int*   dp  = (const int*)  d_in[6];
  const float* dsc = (const float*)d_in[7];
  const int*   sgp = (const int*)  d_in[8];
  const float* sgs = (const float*)d_in[9];
  const int*   sup = (const int*)  d_in[10];
  const float* sus = (const float*)d_in[11];
  const int*   sdp = (const int*)  d_in[12];
  const float* sds = (const float*)d_in[13];
  float* out = (float*)d_out;
  (void)in_sizes; (void)n_in; (void)out_size; (void)ws_size;

  char* w = (char*)d_ws;
  size_t off = 0;
  auto take = [&](size_t bytes) -> char* {
    char* p = w + off;
    off = (off + bytes + 255) & ~(size_t)255;
    return p;
  };
  u16*   xb   = (u16*)  take((size_t)TOK * DDIM * 2);
  int*   tidx = (int*)  take((size_t)TOK * TOPK * 4);
  float* twt  = (float*)take((size_t)TOK * TOPK * 4);
  int*   pos  = (int*)  take((size_t)TOK * TOPK * 4);
  u16*   buf  = (u16*)  take((size_t)NEXP * CAPE * DDIM * 2);
  u16*   Hx   = (u16*)  take((size_t)NEXP * CAPE * IDIM * 2);
  float* eo   = (float*)take((size_t)NEXP * CAPE * DDIM * 4);
  u16*   Hs   = (u16*)  take((size_t)TOK * IDIM * 2);

  // 1) bf16 copy of activations (for shared expert)
  k_to_bf16<<<(TOK * DDIM) / 1024, 256, 0, stream>>>(x, xb);
  // 2) routing
  k_router<<<TOK, NEXP, 0, stream>>>(x, gw, tidx, twt);
  // 3) deterministic slot assignment
  k_assign_pos<<<NEXP, 32, 0, stream>>>(tidx, pos);
  // 4) dispatch tokens into expert buffers
  k_dispatch<<<TOK * TOPK, 256, 0, stream>>>(x, tidx, pos, buf);
  // 5) expert gate/up + SwiGLU  (M=CAP, N=IDIM, K=DDIM)
  k_gemm<DDIM, true><<<dim3(CAPE / 64, IDIM / 64, NEXP), 256, 0, stream>>>(
      buf, gp, gsc, up, usc, Hx, nullptr, IDIM,
      (long)CAPE * DDIM, (long)IDIM * (DDIM / 2), (long)IDIM * (DDIM / GRP),
      (long)CAPE * IDIM);
  // 6) expert down               (M=CAP, N=DDIM, K=IDIM)
  k_gemm<IDIM, false><<<dim3(CAPE / 64, DDIM / 64, NEXP), 256, 0, stream>>>(
      Hx, dp, dsc, nullptr, nullptr, nullptr, eo, DDIM,
      (long)CAPE * IDIM, (long)DDIM * (IDIM / 2), (long)DDIM * (IDIM / GRP),
      (long)CAPE * DDIM);
  // 7) shared expert gate/up + SwiGLU (M=TOK)
  k_gemm<DDIM, true><<<dim3(TOK / 64, IDIM / 64, 1), 256, 0, stream>>>(
      xb, sgp, sgs, sup, sus, Hs, nullptr, IDIM, 0, 0, 0, 0);
  // 8) shared expert down, straight into d_out
  k_gemm<IDIM, false><<<dim3(TOK / 64, DDIM / 64, 1), 256, 0, stream>>>(
      Hs, sdp, sds, nullptr, nullptr, nullptr, out, DDIM, 0, 0, 0, 0);
  // 9) weighted MoE combine on top of shared result
  k_combine<<<TOK, 256, 0, stream>>>(tidx, pos, twt, eo, out);
}
